// MultiHeadAttentionQuantum_65481071396000
// MI455X (gfx1250) — matrix operand order repaired and checked by reading, not compiled, not run on hardware
//
#include <hip/hip_runtime.h>
#include <hip/hip_bf16.h>

// MI455X / gfx1250, wave32. Quantum-proj + flash-attention (WMMA f16 + TDM) + combine.

typedef __attribute__((ext_vector_type(16))) _Float16     v16h;
typedef __attribute__((ext_vector_type(8)))  float        v8f;
typedef __attribute__((ext_vector_type(4)))  unsigned int v4u;
typedef __attribute__((ext_vector_type(8)))  int          v8i;
typedef __attribute__((ext_vector_type(4)))  int          v4i;

#define BATCH   4
#define S_LEN   2048
#define EMB     64
#define NHEAD   8
#define DK      8
#define QT      64              // queries per block (4 M-tiles of 16)
#define CK      32              // keys per LDS chunk
#define NCHUNK  (S_LEN / CK)    // 64
#define NGROUPS (BATCH * S_LEN * NHEAD)   // 65536 rows of 8 wires

#if __has_builtin(__builtin_amdgcn_tensor_load_to_lds) && \
    __has_builtin(__builtin_amdgcn_s_wait_tensorcnt)
#define USE_TDM 1
#else
#define USE_TDM 0
#endif

// ---------------------------------------------------------------------------
// Kernel 1: analytic quantum layer.
// CNOT ring permutes basis states classically ->
// <Z_0> = prod_{j=1..7} cos(x_j+th_j); <Z_w> = prod_{j=0..w} cos(x_j+th_j).
// ---------------------------------------------------------------------------
__global__ __launch_bounds__(256)
void qproj_kernel(const float* __restrict__ x, const float* __restrict__ theta,
                  _Float16* __restrict__ proj) {
  int g = blockIdx.x * blockDim.x + threadIdx.x;
  if (g >= NGROUPS) return;
  const float* xr = x + (size_t)g * 8;
  float C[8];
#pragma unroll
  for (int j = 0; j < 8; ++j) C[j] = __cosf(xr[j] + theta[j]);
  float o[8];
  float t = 1.0f;
#pragma unroll
  for (int j = 1; j < 8; ++j) t *= C[j];
  o[0] = t;                       // all but wire 0
  float p = C[0];
#pragma unroll
  for (int w = 1; w < 8; ++w) { p *= C[w]; o[w] = p; }  // prefix products
  _Float16* dst = proj + (size_t)g * 8;
#pragma unroll
  for (int j = 0; j < 8; ++j) dst[j] = (_Float16)o[j];
}

// ---------------------------------------------------------------------------
// Kernel 2: flash attention (max-free: |score| <= sqrt(8), exp in [0.06,17]).
// Block = 1024 threads = 32 waves; wave w -> head = w>>2, M-tile = w&3.
// kv staged by the Tensor Data Mover; kvT staged by threads in parallel.
// ---------------------------------------------------------------------------
__global__ __launch_bounds__(1024)
void attn_kernel(const _Float16* __restrict__ proj, float* __restrict__ attnO) {
  __shared__ _Float16 kv [CK][EMB];        //  4 KB  [key][feat]  (TDM target)
  __shared__ _Float16 kvT[EMB][CK];        //  4 KB  [feat][key]
  __shared__ _Float16 psc[32][16][CK];     // 32 KB  per-wave P scratch
  __shared__ _Float16 zpad[32];            // 64 B of zeros for padded lanes

  const int tid  = threadIdx.x;
  const int b    = blockIdx.y;
  const int qblk = blockIdx.x;
  const int wave = tid >> 5;
  const int lane = tid & 31;
  const int head = wave >> 2;
  const int mt   = wave & 3;
  const int half = lane >> 4;     // 0: lanes 0-15, 1: lanes 16-31
  const int lrow = lane & 15;
  const int q0   = qblk * QT + mt * 16;

  if (tid < 32) zpad[tid] = (_Float16)0;   // visible after first chunk barrier

  // ---- Q A-fragment (16x32 f16, K=d zero-padded 8->32) -------------------
  // A layout: lanes<16 row M=lrow hold K=0..7 (elts 0..7) & K=16..23 (8..15);
  // lanes>=16 hold K=8..15 & 24..31 -> all padding (zero) here.
  v16h aq;
#pragma unroll
  for (int i = 0; i < 16; ++i) aq[i] = (_Float16)0;
  if (half == 0) {
    const _Float16* qp = proj + ((size_t)(b * S_LEN + q0 + lrow)) * EMB + head * DK;
#pragma unroll
    for (int j = 0; j < 8; ++j) aq[j] = qp[j];
  }

  v8f oacc, sacc;
#pragma unroll
  for (int r = 0; r < 8; ++r) { oacc[r] = 0.0f; sacc[r] = 0.0f; }

  // score B-frag: keep top half (K=16..31 padding) zero across all chunks
  v16h bk;
#pragma unroll
  for (int i = 8; i < 16; ++i) bk[i] = (_Float16)0;

  for (int c = 0; c < NCHUNK; ++c) {
    // ---- stage kvT (transposed) from global; 2 halves per thread ---------
    {
      const int key = tid >> 5;            // 0..31
      const int f   = (tid & 31) * 2;      // 0..62
      const _Float16* src =
          proj + ((size_t)(b * S_LEN + c * CK + key)) * EMB + f;
      _Float16 t0 = src[0], t1 = src[1];
      kvT[f][key]     = t0;  kvT[f + 1][key] = t1;
#if !USE_TDM
      kv[key][f]      = t0;  kv[key][f + 1]  = t1;
#endif
      if (c + 1 < NCHUNK) __builtin_prefetch(src + (size_t)CK * EMB, 0, 1);
    }

#if USE_TDM
    // ---- stage kv via Tensor Data Mover (wave 0 issues one descriptor) ---
    if (wave == 0) {
      unsigned long long ga = (unsigned long long)(uintptr_t)
          (proj + ((size_t)(b * S_LEN + c * CK)) * EMB);
      unsigned lds = (unsigned)(uintptr_t)&kv[0][0];  // addr[31:0] = LDS offset
      // D# group 0: count=1 | lds_addr | global_addr | type=2
      v4u g0 = { 1u, lds, (unsigned)ga,
                 (unsigned)(ga >> 32) | 0x80000000u };
      // D# group 1: data_size=4B; tensor 32x32 dwords; tile 32x32; stride0=32
      v8i g1 = { 0x00020000,          // data_size=2 (4 bytes)
                 (int)(32u << 16),    // tensor_dim0 = 32 dwords (bits 63:48)
                 (int)(32u << 16),    // tensor_dim1 = 32 rows   (bits 95:80)
                 (int)(32u << 16),    // tile_dim0   = 32 dwords (bits 127:112)
                 32,                  // tile_dim1   = 32 rows   (bits 143:128)
                 32,                  // tensor_dim0_stride = 32 dwords
                 0, 0 };
      v4i g2 = { 0, 0, 0, 0 };        // 2-D tile: dims 2/3 unused
      v4i g3 = { 0, 0, 0, 0 };
      v8i g4 = { 0, 0, 0, 0, 0, 0, 0, 0 };  // unused trailing group (clang-23 form)
      __builtin_amdgcn_tensor_load_to_lds(g0, g1, g2, g3, g4, 0);
      __builtin_amdgcn_s_wait_tensorcnt(0);
    }
#endif
    __syncthreads();

    // ---- scores: two 16-key subtiles, K = d padded to 32 -----------------
#pragma unroll
    for (int kt = 0; kt < 2; ++kt) {
      // B layout (32x16): lanes<16 col N=lrow hold K=0..7 (d, elts 0..7) & K=16..23 (8..15),
      // lanes>=16 hold K=8..15 & 24..31 -> padding: read the LDS zero pad instead.
      const _Float16* kp = (half == 0) ? &kv[kt * 16 + lrow][head * DK] : zpad;
#pragma unroll
      for (int j = 0; j < 8; ++j) bk[j] = kp[j];
      v8f cz;
#pragma unroll
      for (int r = 0; r < 8; ++r) cz[r] = 0.0f;
      v8f s = __builtin_amdgcn_wmma_f32_16x16x32_f16(
          false, aq, false, bk, (short)0, cz, false, false);
      // C layout: elt r -> row q = r + 8*half, col = key = lrow (in subtile).
#pragma unroll
      for (int r = 0; r < 8; ++r) {
        float pv = __expf(s[r] * 0.35355339059327373f);  // 1/sqrt(8)
        sacc[r] += pv;
        psc[wave][r + 8 * half][kt * 16 + lrow] = (_Float16)pv;
      }
    }
    __builtin_amdgcn_wave_barrier();  // LDS ops are in-order per wave (DScnt)

    // ---- P (A-frag, C->A via LDS round-trip) x V (B-frag) ----------------
    {
      v16h ap;
      const _Float16* p0 = &psc[wave][lrow][half ? 16 : 0];
      const _Float16* p1 = &psc[wave][lrow][half ? 24 : 8];
#pragma unroll
      for (int j = 0; j < 8; ++j) { ap[j] = p0[j]; ap[8 + j] = p1[j]; }

      // V B-frag (32 keys x 16 n, n = d padded 8->16):
      // lanes<16 col N=lrow keys 0..15; lanes>=16 col N=lrow keys 16..31.
      const _Float16* vp = (lrow < 8) ? &kvT[head * DK + lrow][half * 16] : zpad;
      v16h bv;
#pragma unroll
      for (int j = 0; j < 16; ++j) bv[j] = vp[j];

      oacc = __builtin_amdgcn_wmma_f32_16x16x32_f16(
          false, ap, false, bv, (short)0, oacc, false, false);
    }
    __syncthreads();  // before next chunk overwrites kv/kvT
  }

  // ---- softmax denominator: reduce across the 16 lanes of each half ------
#pragma unroll
  for (int r = 0; r < 8; ++r) {
    float t = sacc[r];
    t += __shfl_xor(t, 1, 32);
    t += __shfl_xor(t, 2, 32);
    t += __shfl_xor(t, 4, 32);
    t += __shfl_xor(t, 8, 32);
    sacc[r] = t;
  }

  // ---- normalize + store: C layout elt r -> (q = r+8*half, d = lrow) -----
  if (lrow < 8) {
#pragma unroll
    for (int r = 0; r < 8; ++r) {
      int q = q0 + r + 8 * half;
      attnO[((size_t)(b * S_LEN + q)) * EMB + head * DK + lrow] =
          oacc[r] / sacc[r];
    }
  }
}

// ---------------------------------------------------------------------------
// Kernel 3: out = O @ W^T  (M=8192, N=64, K=64), f32 VALU for final precision.
// ---------------------------------------------------------------------------
__global__ __launch_bounds__(256)
void combine_kernel(const float* __restrict__ O, const float* __restrict__ W,
                    float* __restrict__ out) {
  __shared__ float Wl[64][65];   // padded vs bank conflicts
  __shared__ float Ol[4][64];
  const int tid = threadIdx.x;
  for (int i = tid; i < 64 * 64; i += 256) Wl[i >> 6][i & 63] = W[i];
  const int row0 = blockIdx.x * 4;
  {
    int i = tid;  // exactly 256 elements
    Ol[i >> 6][i & 63] = O[(size_t)(row0 + (i >> 6)) * 64 + (i & 63)];
  }
  __syncthreads();
  const int r = tid >> 6, e = tid & 63;
  float acc = 0.0f;
#pragma unroll
  for (int f = 0; f < 64; ++f) acc += Ol[r][f] * Wl[e][f];
  out[(size_t)(row0 + r) * 64 + e] = acc;
}

// ---------------------------------------------------------------------------
extern "C" void kernel_launch(void* const* d_in, const int* in_sizes, int n_in,
                              void* d_out, int out_size, void* d_ws, size_t ws_size,
                              hipStream_t stream) {
  const float* x     = (const float*)d_in[0];   // (4,2048,64) f32
  const float* theta = (const float*)d_in[1];   // (8,)        f32
  const float* w     = (const float*)d_in[2];   // (64,64)     f32
  float*       out   = (float*)d_out;           // (4,2048,64) f32

  _Float16* proj  = (_Float16*)d_ws;                                   // 1 MB
  float*    attnO = (float*)((char*)d_ws +
                     (size_t)BATCH * S_LEN * EMB * sizeof(_Float16));  // 2 MB

  qproj_kernel<<<NGROUPS / 256, 256, 0, stream>>>(x, theta, proj);

  dim3 g2(S_LEN / QT, BATCH);   // (32, 4)
  attn_kernel<<<g2, 1024, 0, stream>>>(proj, attnO);

  combine_kernel<<<(BATCH * S_LEN) / 4, 256, 0, stream>>>(attnO, w, out);
}
